// MultiheadCrossAttention_44444321579616
// MI455X (gfx1250) — matrix operand order repaired and checked by reading, not compiled, not run on hardware
//
#include <hip/hip_runtime.h>
#include <hip/hip_bf16.h>

typedef __bf16 bf16_t;
typedef __attribute__((ext_vector_type(16))) __bf16 v16bf;
typedef __attribute__((ext_vector_type(8)))  float  v8f;

#define DIMX      1024
#define NUMH      16
#define HDIM      64
#define BATCH     8
#define NQ        1024
#define NKV       1024

union FragAB { v16bf v; unsigned u[8]; };

__device__ __forceinline__ v8f wmma_bf16(v16bf a, v16bf b, v8f c) {
  return __builtin_amdgcn_wmma_f32_16x16x32_bf16(false, a, false, b, (short)0, c, false, false);
}

// ---------------------------------------------------------------------------
// Kernel 1: fused Q/K/V projection.  C = X @ W^T + b.
// Block tile: 128(M) x 128(N), K-step 32 -> 8 WMMAs per wave per K-step.
// Q,K stored bf16 [B,H,N,64]; V stored TRANSPOSED bf16 [B,H,64,NKV] so the
// attention P@V B-fragments become contiguous dword loads.
// ---------------------------------------------------------------------------
__global__ __launch_bounds__(256) void proj_qkv_kernel(
    const float* __restrict__ query, const float* __restrict__ key_value,
    const float* __restrict__ Wq, const float* __restrict__ bq,
    const float* __restrict__ Wk, const float* __restrict__ bk,
    const float* __restrict__ Wv, const float* __restrict__ bv,
    bf16_t* __restrict__ Qws, bf16_t* __restrict__ Kws, bf16_t* __restrict__ Vws)
{
  __shared__ __align__(16) bf16_t As[128 * 32];
  __shared__ __align__(16) bf16_t Bs[128 * 32];

  const int which = blockIdx.z;
  const float* X    = (which == 0) ? query : key_value;
  const float* W    = (which == 0) ? Wq : (which == 1) ? Wk : Wv;
  const float* bias = (which == 0) ? bq : (which == 1) ? bk : bv;
  bf16_t* dst       = (which == 0) ? Qws : (which == 1) ? Kws : Vws;

  const int m0   = blockIdx.x * 128;
  const int n0   = blockIdx.y * 128;
  const int tid  = threadIdx.x;
  const int lane = tid & 31;
  const int wv   = tid >> 5;        // wave 0..7
  const int hlf  = lane >> 4;       // 0/1 half-wave
  const int l16  = lane & 15;

  v8f acc[8];
  #pragma unroll
  for (int t = 0; t < 8; ++t) acc[t] = v8f{};

  for (int kk = 0; kk < DIMX; kk += 32) {
    // Stage A tile 128x32 (f32 -> bf16): thread -> 16 contiguous elements
    {
      int row = tid >> 1;
      int cb  = (tid & 1) * 16;
      const float* src = X + (size_t)(m0 + row) * DIMX + kk + cb;
      bf16_t* d = As + row * 32 + cb;
      #pragma unroll
      for (int i = 0; i < 16; ++i) d[i] = (bf16_t)src[i];
      if (kk + 32 < DIMX) __builtin_prefetch(src + 32, 0, 2);
    }
    // Stage B tile 128x32: Bs[n][k] = W[n0+n][kk+k]
    {
      int row = tid >> 1;
      int cb  = (tid & 1) * 16;
      const float* src = W + (size_t)(n0 + row) * DIMX + kk + cb;
      bf16_t* d = Bs + row * 32 + cb;
      #pragma unroll
      for (int i = 0; i < 16; ++i) d[i] = (bf16_t)src[i];
      if (kk + 32 < DIMX) __builtin_prefetch(src + 32, 0, 2);
    }
    __syncthreads();

    // A fragment (16x32 bf16): lane m = l16; VGPR i: K=8*hlf+2i, VGPR 4+i: K=16+8*hlf+2i
    FragAB a;
    {
      const bf16_t* ap = As + (wv * 16 + l16) * 32;
      #pragma unroll
      for (int i = 0; i < 4; ++i) {
        a.u[i]     = *(const unsigned*)(ap + hlf * 8 + 2 * i);
        a.u[4 + i] = *(const unsigned*)(ap + 16 + hlf * 8 + 2 * i);
      }
    }
    #pragma unroll
    for (int t = 0; t < 8; ++t) {
      // B fragment (32x16): lane n = l16; VGPR j: K=8*hlf+2j (j<4), K=16+8*hlf+2(j-4) (j>=4)
      FragAB bfr;
      const bf16_t* bp = Bs + (t * 16 + l16) * 32 + hlf * 8;
      #pragma unroll
      for (int j = 0; j < 8; ++j) bfr.u[j] = *(const unsigned*)(bp + 2 * j + (j & 4) * 2);
      acc[t] = wmma_bf16(a.v, bfr.v, acc[t]);
    }
    __syncthreads();
  }

  // C layout: VGPR r, lane -> m = r + 8*hlf, n = l16.
  #pragma unroll
  for (int t = 0; t < 8; ++t) {
    #pragma unroll
    for (int r = 0; r < 8; ++r) {
      int m   = m0 + wv * 16 + r + hlf * 8;
      int c   = n0 + t * 16 + l16;
      int bb  = m >> 10;
      int row = m & 1023;
      int h   = c >> 6;
      int hd  = c & 63;
      float val = acc[t][r] + bias[c];
      if (which == 2) {
        // V transposed: [B,H,64,NKV]
        dst[((size_t)(bb * NUMH + h) * HDIM + hd) * NKV + row] = (bf16_t)val;
      } else {
        dst[((size_t)(bb * NUMH + h) * NQ + row) * HDIM + hd] = (bf16_t)val;
      }
    }
  }
}

// ---------------------------------------------------------------------------
// Kernel 2: attention.  One block = (b, 16-row q tile), loops over 16 heads.
// ---------------------------------------------------------------------------
__global__ __launch_bounds__(256) void attn_kernel(
    const bf16_t* __restrict__ Qws, const bf16_t* __restrict__ Kws,
    const bf16_t* __restrict__ VTws, const float* __restrict__ cluster,
    bf16_t* __restrict__ AO, float* __restrict__ attn_mat)
{
  __shared__ __align__(16) bf16_t Pl[16 * 1032];   // softmax probs, padded stride
  __shared__ float red[16][8];                      // cross-wave row reductions
  __shared__ float ored[4][16][16];                 // P@V partial (K-half 1)

  const int blk = blockIdx.x;
  const int b   = blk >> 6;
  const int q0  = (blk & 63) * 16;
  const int tid  = threadIdx.x;
  const int lane = tid & 31;
  const int wv   = tid >> 5;
  const int hlf  = lane >> 4;
  const int l16  = lane & 15;

  float meanacc[8][8];
  #pragma unroll
  for (int t = 0; t < 8; ++t)
    #pragma unroll
    for (int r = 0; r < 8; ++r) meanacc[t][r] = 0.0f;

  for (int h = 0; h < NUMH; ++h) {
    const bf16_t* Qb  = Qws  + ((size_t)(b * NUMH + h) * NQ + q0) * HDIM;
    const bf16_t* Kb  = Kws  + (size_t)(b * NUMH + h) * NKV * HDIM;
    const bf16_t* VTb = VTws + (size_t)(b * NUMH + h) * HDIM * NKV;  // [64][NKV]

    // Q fragments (16x64 -> two 16x32 A-frags), direct from global (L2-hot).
    FragAB qa[2];
    {
      const bf16_t* qp = Qb + (size_t)l16 * HDIM;
      #pragma unroll
      for (int f = 0; f < 2; ++f)
        #pragma unroll
        for (int i = 0; i < 4; ++i) {
          qa[f].u[i]     = *(const unsigned*)(qp + f * 32 + hlf * 8 + 2 * i);
          qa[f].u[4 + i] = *(const unsigned*)(qp + f * 32 + 16 + hlf * 8 + 2 * i);
        }
    }

    // S = Q @ K^T : each wave owns 128 kv columns (8 tiles), K streamed from L2.
    v8f s[8];
    #pragma unroll
    for (int t = 0; t < 8; ++t) {
      int kvb = wv * 128 + t * 16;
      v8f c = v8f{};
      #pragma unroll
      for (int f = 0; f < 2; ++f) {
        FragAB kb;
        const bf16_t* kp = Kb + (size_t)(kvb + l16) * HDIM + f * 32 + hlf * 8;
        #pragma unroll
        for (int j = 0; j < 8; ++j) kb.u[j] = *(const unsigned*)(kp + 2 * j + (j & 4) * 2);
        c = wmma_bf16(qa[f].v, kb.v, c);
      }
      s[t] = c;
    }

    // scale + cluster bias
    #pragma unroll
    for (int t = 0; t < 8; ++t) {
      int n = wv * 128 + t * 16 + l16;
      #pragma unroll
      for (int r = 0; r < 8; ++r) {
        int m = r + hlf * 8;
        float cm = cluster[((size_t)b * NQ + q0 + m) * NKV + n];
        s[t][r] = s[t][r] * 0.125f + cm;
      }
    }

    // row max
    float rmax[8];
    #pragma unroll
    for (int r = 0; r < 8; ++r) {
      float v = s[0][r];
      #pragma unroll
      for (int t = 1; t < 8; ++t) v = fmaxf(v, s[t][r]);
      #pragma unroll
      for (int off = 8; off >= 1; off >>= 1) v = fmaxf(v, __shfl_xor(v, off, 16));
      rmax[r] = v;
    }
    if (l16 == 0)
      #pragma unroll
      for (int r = 0; r < 8; ++r) red[hlf * 8 + r][wv] = rmax[r];
    __syncthreads();
    #pragma unroll
    for (int r = 0; r < 8; ++r) {
      float v = red[hlf * 8 + r][0];
      #pragma unroll
      for (int w2 = 1; w2 < 8; ++w2) v = fmaxf(v, red[hlf * 8 + r][w2]);
      rmax[r] = v;
    }
    __syncthreads();

    // exp + row sum
    float rinv[8];
    #pragma unroll
    for (int r = 0; r < 8; ++r) {
      float sum = 0.0f;
      #pragma unroll
      for (int t = 0; t < 8; ++t) {
        float p = __expf(s[t][r] - rmax[r]);
        s[t][r] = p;
        sum += p;
      }
      #pragma unroll
      for (int off = 8; off >= 1; off >>= 1) sum += __shfl_xor(sum, off, 16);
      rinv[r] = sum;
    }
    if (l16 == 0)
      #pragma unroll
      for (int r = 0; r < 8; ++r) red[hlf * 8 + r][wv] = rinv[r];
    __syncthreads();
    #pragma unroll
    for (int r = 0; r < 8; ++r) {
      float sum = 0.0f;
      #pragma unroll
      for (int w2 = 0; w2 < 8; ++w2) sum += red[hlf * 8 + r][w2];
      rinv[r] = 1.0f / sum;
    }

    // normalize, accumulate head-mean, stage P (bf16) to LDS [16][1032]
    #pragma unroll
    for (int t = 0; t < 8; ++t) {
      int n = wv * 128 + t * 16 + l16;
      #pragma unroll
      for (int r = 0; r < 8; ++r) {
        int m = r + hlf * 8;
        float p = s[t][r] * rinv[r];
        meanacc[t][r] += p;
        Pl[m * 1032 + n] = (bf16_t)p;
      }
    }
    __syncthreads();

    // attn_out = P @ V : wave -> (n-tile = wv&3, K-half = wv>>2).
    // V^T layout makes each B-fragment VGPR one contiguous dword load.
    {
      int nt    = wv & 3;
      int khalf = wv >> 2;
      const bf16_t* vtp = VTb + (size_t)(nt * 16 + l16) * NKV;
      v8f oacc = v8f{};
      for (int kb = khalf * 512; kb < khalf * 512 + 512; kb += 32) {
        FragAB pa, vb;
        const bf16_t* pp = Pl + l16 * 1032 + kb;
        #pragma unroll
        for (int i = 0; i < 4; ++i) {
          pa.u[i]     = *(const unsigned*)(pp + hlf * 8 + 2 * i);
          pa.u[4 + i] = *(const unsigned*)(pp + 16 + hlf * 8 + 2 * i);
        }
        const bf16_t* vp = vtp + kb + hlf * 8;
        #pragma unroll
        for (int j = 0; j < 8; ++j) vb.u[j] = *(const unsigned*)(vp + 2 * j + (j & 4) * 2);
        oacc = wmma_bf16(pa.v, vb.v, oacc);
      }
      if (khalf == 1) {
        #pragma unroll
        for (int r = 0; r < 8; ++r) ored[nt][r + hlf * 8][l16] = oacc[r];
      }
      __syncthreads();
      if (khalf == 0) {
        #pragma unroll
        for (int r = 0; r < 8; ++r) {
          int m = r + hlf * 8;
          float vfin = oacc[r] + ored[nt][m][l16];
          AO[((size_t)(b * NQ) + q0 + m) * DIMX + h * HDIM + nt * 16 + l16] = (bf16_t)vfin;
        }
      }
      __syncthreads();
    }
  }

  // head-mean attention matrix
  #pragma unroll
  for (int t = 0; t < 8; ++t) {
    int n = wv * 128 + t * 16 + l16;
    #pragma unroll
    for (int r = 0; r < 8; ++r) {
      int m = r + hlf * 8;
      attn_mat[((size_t)b * NQ + q0 + m) * NKV + n] = meanacc[t][r] * (1.0f / 16.0f);
    }
  }
}

// ---------------------------------------------------------------------------
// Kernel 3: output projection.  out = AO(bf16) @ Wo^T + bo, f32 output.
// A tile is a pure bf16 copy -> use CDNA5 async global->LDS DMA path.
// ---------------------------------------------------------------------------
__global__ __launch_bounds__(256) void proj_out_kernel(
    const bf16_t* __restrict__ AO, const float* __restrict__ Wo,
    const float* __restrict__ bo, float* __restrict__ out)
{
  __shared__ __align__(16) bf16_t As[128 * 32];
  __shared__ __align__(16) bf16_t Bs[128 * 32];

  const int m0   = blockIdx.x * 128;
  const int n0   = blockIdx.y * 128;
  const int tid  = threadIdx.x;
  const int lane = tid & 31;
  const int wv   = tid >> 5;
  const int hlf  = lane >> 4;
  const int l16  = lane & 15;

  // Per-thread async-copy addresses: 32 bytes each (2 x b128)
  const int arow = tid >> 1;
  const int acb  = (tid & 1) * 16;
  const unsigned lds_base = (unsigned)(uintptr_t)(As + arow * 32 + acb);
  const unsigned long long gbase = (unsigned long long)AO;

  v8f acc[8];
  #pragma unroll
  for (int t = 0; t < 8; ++t) acc[t] = v8f{};

  for (int kk = 0; kk < DIMX; kk += 32) {
    // Stage A tile 128x32 via async DMA (bf16 copy, no conversion needed)
    {
      unsigned goff = (unsigned)(((size_t)(m0 + arow) * DIMX + kk + acb) * 2);
      asm volatile("global_load_async_to_lds_b128 %0, %1, %2"
                   :: "v"(lds_base), "v"(goff), "s"(gbase) : "memory");
      asm volatile("global_load_async_to_lds_b128 %0, %1, %2"
                   :: "v"(lds_base + 16u), "v"(goff + 16u), "s"(gbase) : "memory");
    }
    // Stage B tile 128x32 (f32 -> bf16)
    {
      const float* src = Wo + (size_t)(n0 + arow) * DIMX + kk + acb;
      bf16_t* d = Bs + arow * 32 + acb;
      #pragma unroll
      for (int i = 0; i < 16; ++i) d[i] = (bf16_t)src[i];
      if (kk + 32 < DIMX) __builtin_prefetch(src + 32, 0, 2);
    }
    asm volatile("s_wait_asynccnt 0x0" ::: "memory");
    __syncthreads();

    FragAB a;
    {
      const bf16_t* ap = As + (wv * 16 + l16) * 32;
      #pragma unroll
      for (int i = 0; i < 4; ++i) {
        a.u[i]     = *(const unsigned*)(ap + hlf * 8 + 2 * i);
        a.u[4 + i] = *(const unsigned*)(ap + 16 + hlf * 8 + 2 * i);
      }
    }
    #pragma unroll
    for (int t = 0; t < 8; ++t) {
      FragAB bfr;
      const bf16_t* bp = Bs + (t * 16 + l16) * 32 + hlf * 8;
      #pragma unroll
      for (int j = 0; j < 8; ++j) bfr.u[j] = *(const unsigned*)(bp + 2 * j + (j & 4) * 2);
      acc[t] = wmma_bf16(a.v, bfr.v, acc[t]);
    }
    __syncthreads();
  }

  #pragma unroll
  for (int t = 0; t < 8; ++t) {
    #pragma unroll
    for (int r = 0; r < 8; ++r) {
      int m = m0 + wv * 16 + r + hlf * 8;
      int c = n0 + t * 16 + l16;
      out[(size_t)m * DIMX + c] = acc[t][r] + bo[c];
    }
  }
}

// ---------------------------------------------------------------------------
extern "C" void kernel_launch(void* const* d_in, const int* in_sizes, int n_in,
                              void* d_out, int out_size, void* d_ws, size_t ws_size,
                              hipStream_t stream) {
  const float* query     = (const float*)d_in[0];
  const float* key_value = (const float*)d_in[1];
  const float* cluster   = (const float*)d_in[2];
  const float* Wq = (const float*)d_in[3];
  const float* bq = (const float*)d_in[4];
  const float* Wk = (const float*)d_in[5];
  const float* bk = (const float*)d_in[6];
  const float* Wv = (const float*)d_in[7];
  const float* bv = (const float*)d_in[8];
  const float* Wo = (const float*)d_in[9];
  const float* bo = (const float*)d_in[10];

  const size_t elems = (size_t)BATCH * NQ * DIMX;  // 8M
  bf16_t* Qws = (bf16_t*)d_ws;
  bf16_t* Kws = Qws + elems;
  bf16_t* Vws = Kws + elems;   // V stored transposed per head: [B,H,64,NKV]
  bf16_t* AO  = Vws + elems;

  float* out_q    = (float*)d_out;
  float* out_attn = out_q + elems;

  dim3 gproj(BATCH * NQ / 128, DIMX / 128, 3);
  proj_qkv_kernel<<<gproj, 256, 0, stream>>>(query, key_value, Wq, bq, Wk, bk,
                                             Wv, bv, Qws, Kws, Vws);

  attn_kernel<<<dim3(BATCH * NQ / 16), 256, 0, stream>>>(Qws, Kws, Vws, cluster,
                                                         AO, out_attn);

  proj_out_kernel<<<dim3(BATCH * NQ / 128, DIMX / 128), 256, 0, stream>>>(AO, Wo,
                                                                          bo, out_q);
}
